// HyperNet_40114994545045
// MI455X (gfx1250) — matrix operand order repaired and checked by reading, not compiled, not run on hardware
//
#include <hip/hip_runtime.h>

// ---------------------------------------------------------------------------
// HyperNet forward for MI455X (gfx1250): bf16 WMMA implicit-GEMM convolutions
// with Tensor-Data-Mover (TDM) staging of strided 2-D tiles into LDS.
// Pipeline:
//   1) cvt_bf16        : x f32 -> bf16
//   2) hypergen        : per-sample 5x5 weights (bf16) + per-sample bias (f32)
//   3) conv5x5_wmma    : per-sample implicit GEMM, M=96, N=4096, K=2400
//                        A slab (96x32, stride 2400) staged via tensor_load_to_lds
//   4) conv1x1_wmma x2 : GEMM M=96, K=96, N=4096/sample; X slab staged via TDM
//   5) pool3x3s2       : 3x3 stride-2 sum/9 with pad 1 -> f32 output
// ---------------------------------------------------------------------------

typedef __bf16 bf16_t;
typedef __bf16 v16bf __attribute__((ext_vector_type(16)));
typedef float  v8f   __attribute__((ext_vector_type(8)));
typedef unsigned int v4u __attribute__((ext_vector_type(4)));
typedef int v8i __attribute__((ext_vector_type(8)));
typedef int v4i __attribute__((ext_vector_type(4)));

#define B_N    64
#define C_N    96
#define HW     4096
#define K5     2400      // 96 * 25
#define WSIZE  230400    // 96*96*5*5
#define TOTN   230496    // WSIZE + 96

#define LDA 36           // 32-elem rows + 2-DW TDM pad  (64B rows, interval=3, amount=1)
#define LDB 268          // 256-elem rows + 6-DW TDM pad (512B rows, interval=6, amount=5)

// ---- issue one 2-D TDM tile load: rows of tile_d0 bf16, tile_d1 rows -------
// data_size = 2 bytes. Descriptor per CDNA5 ISA ch. 8 (groups 0/1; 2D => g2/g3 zero).
// This toolchain exposes the 6-arg builtin: (u32x4, i32x8, i32x4, i32x4, i32x8, cpol).
__device__ __forceinline__ void tdm_load_2d_bf16(unsigned lds_addr, const void* gptr,
                                                 unsigned tile_d0, unsigned tile_d1,
                                                 unsigned tensor_d0, unsigned tensor_d1,
                                                 unsigned long long stride0,
                                                 unsigned pad_interval, unsigned pad_amount) {
  unsigned long long ga = (unsigned long long)gptr;
  v4u g0;
  g0.x = 1u;                                                // count=1 (valid), user mode
  g0.y = lds_addr;                                          // LDS byte address
  g0.z = (unsigned)ga;                                      // global_addr[31:0]
  g0.w = (unsigned)((ga >> 32) & 0x01FFFFFFu) | (2u << 30); // global_addr[56:32] | type=2
  v8i g1;
  g1[0] = (int)((1u << 16)                 // data_size = 1 -> 2 bytes
              | (1u << 20)                 // pad_enable
              | (pad_interval << 22)
              | (pad_amount << 25));
  g1[1] = (int)((tensor_d0 & 0xFFFFu) << 16);                       // tensor_dim0[15:0]
  g1[2] = (int)((tensor_d0 >> 16) | ((tensor_d1 & 0xFFFFu) << 16)); // dim0 hi | dim1 lo
  g1[3] = (int)((tensor_d1 >> 16) | (tile_d0 << 16));               // dim1 hi | tile_dim0
  g1[4] = (int)(tile_d1 & 0xFFFFu);                                 // tile_dim1 (tile_dim2=0)
  g1[5] = (int)(unsigned)(stride0 & 0xFFFFFFFFull);                 // tensor_dim0_stride lo
  g1[6] = (int)(unsigned)(stride0 >> 32);                           // stride hi (dim1_stride=0)
  g1[7] = 0;
  v4i z4 = {0, 0, 0, 0};
  v8i z8 = {0, 0, 0, 0, 0, 0, 0, 0};
  __builtin_amdgcn_tensor_load_to_lds(g0, g1, z4, z4, z8, 0);
}

// ---------------- elementwise: f32 -> bf16 ----------------
__global__ __launch_bounds__(256) void cvt_bf16_kernel(const float* __restrict__ in,
                                                       bf16_t* __restrict__ out, int n) {
  int t = blockIdx.x * 256 + threadIdx.x;
  if (t < n) out[t] = (bf16_t)in[t];
}

// ---------------- hypernetwork weight/bias generation ----------------
__global__ __launch_bounds__(256) void hypergen_kernel(const float* __restrict__ h_in,
                                                       const float* __restrict__ lin_w,
                                                       const float* __restrict__ lin_b,
                                                       bf16_t* __restrict__ wts,
                                                       float* __restrict__ biasbuf) {
  long t = (long)blockIdx.x * 256 + threadIdx.x;
  if (t >= (long)B_N * TOTN) return;
  int b = (int)(t / TOTN);
  int i = (int)(t - (long)b * TOTN);
  float hv = h_in[b];
  float h  = (hv > 0.0001f ? 0.5f : 0.0f) + hv * (1.0f / 64.0f);
  float v  = h * lin_w[i] + lin_b[i];
  if (i < WSIZE) wts[(long)b * WSIZE + i] = (bf16_t)v;     // [b][o][cin*25+ky*5+kx]
  else           biasbuf[b * C_N + (i - WSIZE)] = v;       // [b][o]
}

// ---------------- per-sample 5x5 conv as implicit GEMM (WMMA bf16) ----------
// grid = (16 pixel-tiles of 256, 64 samples), block = 256 (8 waves x 32 cols)
__global__ __launch_bounds__(256) void conv5x5_wmma_kernel(const bf16_t* __restrict__ xb,
                                                           const bf16_t* __restrict__ wts,
                                                           const float*  __restrict__ biasbuf,
                                                           bf16_t* __restrict__ y1) {
  __shared__ bf16_t lA[C_N * LDA];   // A slab: 96 (M) x 32 (K), TDM-padded rows
  __shared__ bf16_t lB[32 * LDB];    // B slab: 32 (K) x 256 (N), padded rows

  const int b    = blockIdx.y;
  const int pix0 = blockIdx.x * 256;
  const int t    = threadIdx.x;
  const int wave = t >> 5, lane = t & 31, lh = lane & 15;
  const int akb  = (lane < 16) ? 0 : 16;   // A-fragment K base (contiguous k = 16*h + e)
  const int bkb  = (lane < 16) ? 0 : 16;   // B-fragment K base (contiguous k = 16*h + e)

  v8f acc0[6], acc1[6];
  #pragma unroll
  for (int i = 0; i < 6; ++i)
    #pragma unroll
    for (int r = 0; r < 8; ++r) { acc0[i][r] = 0.0f; acc1[i][r] = 0.0f; }

  const bf16_t* wbase = wts + (long)b * (long)C_N * K5;
  const bf16_t* xbase = xb  + ((long)b * C_N << 12);
  const unsigned ldsA = (unsigned)(size_t)&lA[0];

  for (int kk = 0; kk < K5; kk += 32) {
    // ---- TDM: DMA the 96x32 weight slab (row stride 2400) into lA ----
    if (wave == 0) {
      tdm_load_2d_bf16(ldsA, wbase + kk,
                       /*tile*/ 32, 96, /*tensor*/ (unsigned)K5, 96,
                       /*stride0*/ (unsigned long long)K5,
                       /*pad_interval(64B)*/ 3, /*pad_amount(2DW)*/ 1);
    }
    // ---- stage B: 32x256 im2col patch (one K row = fixed cin,ky,kx) ----
    {
      int j = t >> 3, nb = (t & 7) * 32;
      int k = kk + j;
      int cin = k / 25, r = k - cin * 25;
      int ky = r / 5, kx = r - ky * 5;
      const bf16_t* xc = xbase + ((long)cin << 12);
      #pragma unroll
      for (int q = 0; q < 32; ++q) {
        int n  = nb + q, p = pix0 + n;
        int py = p >> 6, px = p & 63;
        int iy = py + ky - 2, ix = px + kx - 2;
        bf16_t v = (bf16_t)0.0f;
        if (iy >= 0 && iy < 64 && ix >= 0 && ix < 64) v = xc[(iy << 6) + ix];
        lB[j * LDB + n] = v;
      }
    }
    if (wave == 0) __builtin_amdgcn_s_wait_tensorcnt(0);
    __syncthreads();

    // ---- fragments + 12 WMMAs per wave per K-step ----
    const int nw = wave * 32;
    v16bf bf0, bf1;
    #pragma unroll
    for (int e = 0; e < 16; ++e) {
      bf0[e] = lB[(bkb + e) * LDB + nw + lh];
      bf1[e] = lB[(bkb + e) * LDB + nw + 16 + lh];
    }
    #pragma unroll
    for (int i = 0; i < 6; ++i) {
      v16bf af;
      #pragma unroll
      for (int e = 0; e < 16; ++e) {
        int kidx = akb + e;
        af[e] = lA[(i * 16 + lh) * LDA + kidx];
      }
      acc0[i] = __builtin_amdgcn_wmma_f32_16x16x32_bf16(
          false, af, false, bf0, (short)0, acc0[i], false, false);
      acc1[i] = __builtin_amdgcn_wmma_f32_16x16x32_bf16(
          false, af, false, bf1, (short)0, acc1[i], false, false);
    }
    __syncthreads();
  }

  // ---- epilogue: +bias, ReLU, store bf16 (C layout: lane=N, vgpr r=M row) --
  const int p0   = pix0 + wave * 32 + lh;
  const int mofs = (lane < 16) ? 0 : 8;
  #pragma unroll
  for (int i = 0; i < 6; ++i)
    #pragma unroll
    for (int r = 0; r < 8; ++r) {
      int o = i * 16 + r + mofs;
      float bias = biasbuf[b * C_N + o];
      float v0 = acc0[i][r] + bias; v0 = v0 > 0.0f ? v0 : 0.0f;
      float v1 = acc1[i][r] + bias; v1 = v1 > 0.0f ? v1 : 0.0f;
      long base = ((long)(b * C_N + o) << 12);
      y1[base + p0]      = (bf16_t)v0;
      y1[base + p0 + 16] = (bf16_t)v1;
    }
}

// ---------------- 1x1 conv (96->96 GEMM) + bias + ReLU (WMMA bf16) ----------
#define LDW 100
__global__ __launch_bounds__(256) void conv1x1_wmma_kernel(const bf16_t* __restrict__ in,
                                                           const float* __restrict__ wfp,
                                                           const float* __restrict__ bias,
                                                           bf16_t* __restrict__ out) {
  __shared__ bf16_t lW[C_N * LDW];   // 96 x 96 weights (bf16)
  __shared__ bf16_t lX[32 * LDB];    // 32 (K) x 256 (N), TDM-padded rows

  const int b    = blockIdx.y;
  const int pix0 = blockIdx.x * 256;
  const int t    = threadIdx.x;
  const int wave = t >> 5, lane = t & 31, lh = lane & 15;
  const int akb  = (lane < 16) ? 0 : 16;
  const int bkb  = (lane < 16) ? 0 : 16;

  // stage + convert all 96x96 weights once
  #pragma unroll
  for (int q = 0; q < 36; ++q) {
    int idx = t * 36 + q;
    int m = idx / 96, k = idx - 96 * m;
    lW[m * LDW + k] = (bf16_t)wfp[idx];
  }

  v8f acc0[6], acc1[6];
  #pragma unroll
  for (int i = 0; i < 6; ++i)
    #pragma unroll
    for (int r = 0; r < 8; ++r) { acc0[i][r] = 0.0f; acc1[i][r] = 0.0f; }

  const bf16_t* ibase = in + ((long)b * C_N << 12) + pix0;
  const unsigned ldsX = (unsigned)(size_t)&lX[0];

  for (int kk = 0; kk < 96; kk += 32) {
    // ---- TDM: DMA 32 channels x 256 pixels (row stride 4096) into lX ----
    if (wave == 0) {
      tdm_load_2d_bf16(ldsX, ibase + ((long)kk << 12),
                       /*tile*/ 256, 32, /*tensor*/ (unsigned)HW, 96,
                       /*stride0*/ (unsigned long long)HW,
                       /*pad_interval(512B)*/ 6, /*pad_amount(6DW)*/ 5);
      __builtin_amdgcn_s_wait_tensorcnt(0);
    }
    __syncthreads();   // first iteration also publishes lW

    const int nw = wave * 32;
    v16bf bf0, bf1;
    #pragma unroll
    for (int e = 0; e < 16; ++e) {
      bf0[e] = lX[(bkb + e) * LDB + nw + lh];
      bf1[e] = lX[(bkb + e) * LDB + nw + 16 + lh];
    }
    #pragma unroll
    for (int i = 0; i < 6; ++i) {
      v16bf af;
      #pragma unroll
      for (int e = 0; e < 16; ++e) {
        int kidx = kk + akb + e;
        af[e] = lW[(i * 16 + lh) * LDW + kidx];
      }
      acc0[i] = __builtin_amdgcn_wmma_f32_16x16x32_bf16(
          false, af, false, bf0, (short)0, acc0[i], false, false);
      acc1[i] = __builtin_amdgcn_wmma_f32_16x16x32_bf16(
          false, af, false, bf1, (short)0, acc1[i], false, false);
    }
    __syncthreads();
  }

  const int p0   = pix0 + wave * 32 + lh;
  const int mofs = (lane < 16) ? 0 : 8;
  #pragma unroll
  for (int i = 0; i < 6; ++i)
    #pragma unroll
    for (int r = 0; r < 8; ++r) {
      int o = i * 16 + r + mofs;
      float bv = bias[o];
      float v0 = acc0[i][r] + bv; v0 = v0 > 0.0f ? v0 : 0.0f;
      float v1 = acc1[i][r] + bv; v1 = v1 > 0.0f ? v1 : 0.0f;
      long base = ((long)(b * C_N + o) << 12);
      out[base + p0]      = (bf16_t)v0;
      out[base + p0 + 16] = (bf16_t)v1;
    }
}

// ---------------- 3x3 stride-2 sum/9 pool (pad 1) -> f32 ----------------
__global__ __launch_bounds__(256) void pool_kernel(const bf16_t* __restrict__ y,
                                                   float* __restrict__ outp) {
  int t  = blockIdx.x * 256 + threadIdx.x;     // 64*96*32*32 total
  int ox = t & 31, oy = (t >> 5) & 31;
  int bc = t >> 10;
  const bf16_t* src = y + ((long)bc << 12);
  float s = 0.0f;
  #pragma unroll
  for (int dy = 0; dy < 3; ++dy) {
    int iy = oy * 2 - 1 + dy;
    if (iy < 0 || iy >= 64) continue;
    #pragma unroll
    for (int dx = 0; dx < 3; ++dx) {
      int ix = ox * 2 - 1 + dx;
      if (ix < 0 || ix >= 64) continue;
      s += (float)src[(iy << 6) + ix];
    }
  }
  outp[t] = s * (1.0f / 9.0f);
}

// ---------------------------------------------------------------------------
extern "C" void kernel_launch(void* const* d_in, const int* in_sizes, int n_in,
                              void* d_out, int out_size, void* d_ws, size_t ws_size,
                              hipStream_t stream) {
  (void)in_sizes; (void)n_in; (void)out_size; (void)ws_size;

  const float* x     = (const float*)d_in[0];
  const float* h_in  = (const float*)d_in[1];
  const float* lin_w = (const float*)d_in[2];
  const float* lin_b = (const float*)d_in[3];
  const float* c1w   = (const float*)d_in[4];
  const float* c1b   = (const float*)d_in[5];
  const float* c2w   = (const float*)d_in[6];
  const float* c2b   = (const float*)d_in[7];

  char* ws = (char*)d_ws;
  size_t off = 0;
  bf16_t* xb  = (bf16_t*)(ws + off); off += (size_t)B_N * C_N * HW * 2;   // 50.3 MB
  bf16_t* wts = (bf16_t*)(ws + off); off += (size_t)B_N * WSIZE * 2;      // 29.5 MB
  float*  bb  = (float*) (ws + off); off += (size_t)B_N * C_N * 4;
  bf16_t* y1  = (bf16_t*)(ws + off); off += (size_t)B_N * C_N * HW * 2;   // 50.3 MB
  bf16_t* y2  = (bf16_t*)(ws + off); off += (size_t)B_N * C_N * HW * 2;   // 50.3 MB
  bf16_t* y3  = xb;  // x_bf16 dead after conv5x5 -> reuse for last conv output

  int nx = B_N * C_N * HW;                                   // 25,165,824
  cvt_bf16_kernel<<<(nx + 255) / 256, 256, 0, stream>>>(x, xb, nx);

  long nh = (long)B_N * TOTN;                                // 14,751,744
  hypergen_kernel<<<(int)((nh + 255) / 256), 256, 0, stream>>>(h_in, lin_w, lin_b, wts, bb);

  dim3 g(16, B_N);                                           // 16 pixel tiles x 64 samples
  conv5x5_wmma_kernel<<<g, 256, 0, stream>>>(xb, wts, bb, y1);
  conv1x1_wmma_kernel<<<g, 256, 0, stream>>>(y1, c1w, c1b, y2);
  conv1x1_wmma_kernel<<<g, 256, 0, stream>>>(y2, c2w, c2b, y3);

  int np = B_N * C_N * 32 * 32;                              // 6,291,456
  pool_kernel<<<np / 256, 256, 0, stream>>>(y3, (float*)d_out);
}
